// PPTAttention_21792664060026
// MI455X (gfx1250) — matrix operand order repaired and checked by reading, not compiled, not run on hardware
//
#include <hip/hip_runtime.h>
#include <math.h>

// ---------------- CDNA5 WMMA / TDM types ----------------
typedef __attribute__((ext_vector_type(16))) __bf16 v16bf;
typedef __attribute__((ext_vector_type(8)))  float  v8f;
typedef unsigned int u32x4 __attribute__((ext_vector_type(4)));
typedef int          i32x4 __attribute__((ext_vector_type(4)));
typedef int          i32x8 __attribute__((ext_vector_type(8)));

#define DIMC   768
#define NTOK   577
#define BATCH  16
#define HEADS  12
#define HD     64
#define NP     608                 // padded sequence length (multiple of 32)
#define MROWS  (BATCH * NTOK)      // 9232
#define MPAD   (BATCH * NP)        // 9728 (proj GEMM padded M)
#define QKVC   (3 * DIMC)          // 2304
#define KITERS (DIMC / 32)         // 24
#define SCALEF 0.125f              // 64^-0.5
#define CLSCST 9999.0f

#ifndef PPT_HAS_TDM
#if defined(__has_builtin)
#if __has_builtin(__builtin_amdgcn_tensor_load_to_lds)
#define PPT_HAS_TDM 1
#endif
#endif
#endif
#ifndef PPT_HAS_TDM
#define PPT_HAS_TDM 0
#endif

union FragB16 {
    v16bf v;
    unsigned short u[16];
    unsigned int   w[8];
};

__device__ __forceinline__ unsigned short f2bf(float f) {
    unsigned int u = __float_as_uint(f);
    u += 0x7FFFu + ((u >> 16) & 1u);   // round-to-nearest-even
    return (unsigned short)(u >> 16);
}
__device__ __forceinline__ float bf2f(unsigned short h) {
    return __uint_as_float(((unsigned int)h) << 16);
}

#if PPT_HAS_TDM
// D# group1 for a 2D bf16 tile (tile_d0 x tile_d1) from a (tensor_d1 x
// tensor_d0) row-major tensor. LDS padding: +16B after every 64B stored
// (pad_interval=3, pad_amount=3) -> 40-short LDS row stride for a 32-elem row.
__device__ __forceinline__ i32x8 tdm_desc2d(int tensor_d0, int tensor_d1,
                                            int tile_d0, int tile_d1,
                                            int stride0) {
    i32x8 g;
    g[0] = (int)((1u << 16)          // data_size = 2 bytes
               | (1u << 20)          // pad_enable
               | (3u << 22)          // pad_interval: 16 dwords (64B)
               | (3u << 25));        // pad_amount:   4 dwords (16B)
    g[1] = (int)(((unsigned)tensor_d0 & 0xFFFFu) << 16);
    g[2] = (int)((((unsigned)tensor_d0 >> 16) & 0xFFFFu) |
                 (((unsigned)tensor_d1 & 0xFFFFu) << 16));
    g[3] = (int)((((unsigned)tensor_d1 >> 16) & 0xFFFFu) |
                 (((unsigned)tile_d0 & 0xFFFFu) << 16));
    g[4] = (int)((unsigned)tile_d1 & 0xFFFFu);   // tile_dim1, tile_dim2=0
    g[5] = stride0;                              // tensor_dim0_stride lo
    g[6] = 0;
    g[7] = 0;
    return g;
}

__device__ __forceinline__ void tdm_issue(unsigned lds_off,
                                          unsigned long long gaddr,
                                          i32x8 g1) {
    u32x4 g0;
    g0.x = 1u;                                         // count=1 (valid user D#)
    g0.y = lds_off;                                    // lds_addr (bytes)
    g0.z = (unsigned)(gaddr & 0xFFFFFFFFull);          // global_addr[31:0]
    g0.w = (unsigned)((gaddr >> 32) & 0x1FFFFFFull)    // global_addr[56:32]
         | 0x80000000u;                                // type=2 ("image")
    i32x4 z4 = {0, 0, 0, 0};
#if defined(__clang_major__) && (__clang_major__ >= 23)
    i32x8 z8 = {0, 0, 0, 0, 0, 0, 0, 0};
    __builtin_amdgcn_tensor_load_to_lds(g0, g1, z4, z4, z8, 0);
#else
    __builtin_amdgcn_tensor_load_to_lds(g0, g1, z4, z4, 0);
#endif
}
#endif  // PPT_HAS_TDM

// ---------------- fp32 -> bf16 conversion (vectorized) ----------------
__global__ void ppt_cvt_bf16(const float* __restrict__ src,
                             unsigned short* __restrict__ dst, int n4) {
    for (int i = blockIdx.x * blockDim.x + threadIdx.x; i < n4;
         i += gridDim.x * blockDim.x) {
        const float4 f = ((const float4*)src)[i];
        ushort4 o;
        o.x = f2bf(f.x); o.y = f2bf(f.y); o.z = f2bf(f.z); o.w = f2bf(f.w);
        ((ushort4*)dst)[i] = o;
    }
}

// ---------------- QKV GEMM: (9232 x 768) @ (2304 x 768)^T ----------------
// WG tile 128(M) x 64(N); 8 waves in 4x2; each wave a 2x2 block of WMMA tiles.
// A/B tiles DMA'd by the Tensor Data Mover, double buffered.
__global__ __launch_bounds__(256)
void ppt_qkv_gemm(const unsigned short* __restrict__ xb,
                  const unsigned short* __restrict__ wb,
                  unsigned short* __restrict__ Qb,
                  unsigned short* __restrict__ Kb,
                  unsigned short* __restrict__ Vb) {
    __shared__ unsigned short As[2][128 * 40];
    __shared__ unsigned short Bs[2][64 * 40];
    const int t    = threadIdx.x;
    const int lane = t & 31;
    const int wave = t >> 5;
    const int half = lane >> 4;
    const int l15  = lane & 15;
    const int m0   = blockIdx.x * 128;
    const int n0   = blockIdx.y * 64;
    const int mw   = (wave & 3) * 32;
    const int nw   = (wave >> 2) * 32;

    const v8f vzero = {0.f, 0.f, 0.f, 0.f, 0.f, 0.f, 0.f, 0.f};
    v8f acc[2][2];
#pragma unroll
    for (int i = 0; i < 2; ++i)
#pragma unroll
        for (int j = 0; j < 2; ++j) acc[i][j] = vzero;

#if PPT_HAS_TDM
    const i32x8 gA = tdm_desc2d(DIMC, MROWS, 32, 128, DIMC);  // OOB rows -> 0
    const i32x8 gB = tdm_desc2d(DIMC, QKVC, 32, 64, DIMC);
    const unsigned long long baseA =
        (unsigned long long)(uintptr_t)xb + (unsigned long long)m0 * DIMC * 2;
    const unsigned long long baseB =
        (unsigned long long)(uintptr_t)wb + (unsigned long long)n0 * DIMC * 2;
    if (wave == 0) {
        tdm_issue((unsigned)(uintptr_t)&As[0][0], baseA, gA);
        tdm_issue((unsigned)(uintptr_t)&Bs[0][0], baseB, gB);
    }
#endif

    for (int kt = 0; kt < KITERS; ++kt) {
        const int cur = kt & 1;
#if PPT_HAS_TDM
        if (wave == 0) {
            if (kt + 1 < KITERS) {   // prefetch next K-tile via TDM
                tdm_issue((unsigned)(uintptr_t)&As[cur ^ 1][0],
                          baseA + (unsigned long long)(kt + 1) * 64, gA);
                tdm_issue((unsigned)(uintptr_t)&Bs[cur ^ 1][0],
                          baseB + (unsigned long long)(kt + 1) * 64, gB);
                __builtin_amdgcn_s_wait_tensorcnt(2);  // current pair done
            } else {
                __builtin_amdgcn_s_wait_tensorcnt(0);
            }
        }
        __syncthreads();
#else
        {   // cooperative load fallback
            const int k0 = kt * 32;
            const int r = t >> 2, c = (t & 3) * 8;
#pragma unroll
            for (int rr = 0; rr < 2; ++rr) {
                const int row = m0 + r + rr * 64;
                uint4 val = make_uint4(0u, 0u, 0u, 0u);
                if (row < MROWS)
                    val = *(const uint4*)(xb + (size_t)row * DIMC + k0 + c);
                *(uint4*)(&As[cur][(r + rr * 64) * 40 + c]) = val;
            }
            *(uint4*)(&Bs[cur][r * 40 + c]) =
                *(const uint4*)(wb + (size_t)(n0 + r) * DIMC + k0 + c);
        }
        __syncthreads();
#endif
        FragB16 a[2], b[2];
        const int kb  = half * 8;
        const int kb2 = half * 16;
#pragma unroll
        for (int i = 0; i < 2; ++i) {
            const int m = mw + i * 16 + l15;
#pragma unroll
            for (int g = 0; g < 8; ++g) {
                const int K = kb2 + 2 * g;
                a[i].w[g] = *(const unsigned int*)(&As[cur][m * 40 + K]);
            }
            const int n = nw + i * 16 + l15;
#pragma unroll
            for (int g = 0; g < 8; ++g)
                b[i].w[g] = *(const unsigned int*)(&Bs[cur][n * 40 + kb2 + 2 * g]);
        }
#pragma unroll
        for (int i = 0; i < 2; ++i)
#pragma unroll
            for (int j = 0; j < 2; ++j)
                acc[i][j] = __builtin_amdgcn_wmma_f32_16x16x32_bf16(
                    false, a[i].v, false, b[j].v, (short)0, acc[i][j],
                    false, false);
        __syncthreads();
    }

    // epilogue: scatter into Q/K/V (B,H,NP,HD) bf16 buffers
#pragma unroll
    for (int j = 0; j < 2; ++j) {
        const int col   = n0 + nw + j * 16 + l15;   // 0..2303
        const int which = col / DIMC;
        const int rem   = col % DIMC;
        const int h     = rem >> 6;
        const int d     = rem & 63;
        unsigned short* outp = (which == 0) ? Qb : ((which == 1) ? Kb : Vb);
#pragma unroll
        for (int i = 0; i < 2; ++i)
#pragma unroll
            for (int r = 0; r < 8; ++r) {
                const int row = m0 + mw + i * 16 + r + half * 8;
                if (row < MROWS) {
                    const int bb = row / NTOK, nn = row % NTOK;
                    outp[(size_t)((bb * HEADS + h) * NP + nn) * HD + d] =
                        f2bf(acc[i][j][r]);
                }
            }
    }
}

// ---------------- fused flash attention ----------------
__global__ __launch_bounds__(32)
void ppt_flash_attn(const unsigned short* __restrict__ Qb,
                    const unsigned short* __restrict__ Kb,
                    const unsigned short* __restrict__ Vb,
                    unsigned short* __restrict__ AOb) {
    __shared__ unsigned short Pl[16 * 36];
    const int lane = threadIdx.x & 31;
    const int half = lane >> 4;
    const int l15  = lane & 15;
    const int bh   = blockIdx.y;
    const int qt   = blockIdx.x;
    const int bb   = bh / HEADS, hh = bh % HEADS;

    const unsigned short* Qrow = Qb + (size_t)bh * NP * HD;
    const unsigned short* Krow = Kb + (size_t)bh * NP * HD;
    const unsigned short* Vrow = Vb + (size_t)bh * NP * HD;

    FragB16 qa[2];
    {
        const int qrow = qt * 16 + l15;
        const int kb   = half * 8;
#pragma unroll
        for (int ch = 0; ch < 2; ++ch)
#pragma unroll
            for (int g = 0; g < 8; ++g) {
                const int K = 2 * kb + 2 * g;
                qa[ch].w[g] = *(const unsigned int*)(
                    Qrow + (size_t)qrow * HD + ch * 32 + K);
            }
    }

    v8f o[4];
    const v8f vzero = {0.f, 0.f, 0.f, 0.f, 0.f, 0.f, 0.f, 0.f};
#pragma unroll
    for (int i = 0; i < 4; ++i) o[i] = vzero;
    float m_r[8], l_r[8];
#pragma unroll
    for (int r = 0; r < 8; ++r) { m_r[r] = -1e30f; l_r[r] = 0.f; }

    for (int kt = 0; kt < 19; ++kt) {     // 19 * 32 = 608 keys
        const int key0 = kt * 32;
        v8f s0 = vzero, s1 = vzero;
#pragma unroll
        for (int ch = 0; ch < 2; ++ch) {
            FragB16 kf0, kf1;
            const int kd = ch * 32 + half * 16;
#pragma unroll
            for (int g = 0; g < 8; ++g) {
                kf0.w[g] = *(const unsigned int*)(
                    Krow + (size_t)(key0 + l15) * HD + kd + 2 * g);
                kf1.w[g] = *(const unsigned int*)(
                    Krow + (size_t)(key0 + 16 + l15) * HD + kd + 2 * g);
            }
            s0 = __builtin_amdgcn_wmma_f32_16x16x32_bf16(
                false, qa[ch].v, false, kf0.v, (short)0, s0, false, false);
            s1 = __builtin_amdgcn_wmma_f32_16x16x32_bf16(
                false, qa[ch].v, false, kf1.v, (short)0, s1, false, false);
        }
        const bool c0 = (key0 + l15) < NTOK;
        const bool c1 = (key0 + 16 + l15) < NTOK;
#pragma unroll
        for (int r = 0; r < 8; ++r) {
            float a = c0 ? s0[r] * SCALEF : -1e30f;
            float c = c1 ? s1[r] * SCALEF : -1e30f;
            float t = fmaxf(a, c);
#pragma unroll
            for (int x = 8; x >= 1; x >>= 1)   // xor<=8 stays inside the half
                t = fmaxf(t, __shfl_xor(t, x, 32));
            const float mn    = fmaxf(m_r[r], t);
            const float alpha = __expf(m_r[r] - mn);
            const float p0    = __expf(a - mn);
            const float p1    = __expf(c - mn);
            float rs = p0 + p1;
#pragma unroll
            for (int x = 8; x >= 1; x >>= 1) rs += __shfl_xor(rs, x, 32);
            l_r[r] = l_r[r] * alpha + rs;
            m_r[r] = mn;
#pragma unroll
            for (int i = 0; i < 4; ++i) o[i][r] *= alpha;
            const int prow = r + half * 8;
            Pl[prow * 36 + l15]      = f2bf(p0);
            Pl[prow * 36 + 16 + l15] = f2bf(p1);
        }
        __syncthreads();
        FragB16 pf;
        {
            const int kb = half * 8;
#pragma unroll
            for (int g = 0; g < 8; ++g) {
                const int K = 2 * kb + 2 * g;
                pf.w[g] = *(const unsigned int*)(&Pl[l15 * 36 + K]);
            }
        }
#pragma unroll
        for (int ns = 0; ns < 4; ++ns) {
            FragB16 vf;
            const int kb2 = half * 16;
#pragma unroll
            for (int jj = 0; jj < 16; ++jj)
                vf.u[jj] = Vrow[(size_t)(key0 + kb2 + jj) * HD + ns * 16 + l15];
            o[ns] = __builtin_amdgcn_wmma_f32_16x16x32_bf16(
                false, pf.v, false, vf.v, (short)0, o[ns], false, false);
        }
        __syncthreads();
    }

#pragma unroll
    for (int ns = 0; ns < 4; ++ns)
#pragma unroll
        for (int r = 0; r < 8; ++r) {
            const int row = qt * 16 + r + half * 8;
            if (row < NTOK) {
                const float val = o[ns][r] / l_r[r];
                AOb[(size_t)(bb * NP + row) * DIMC + hh * HD + ns * 16 + l15] =
                    f2bf(val);
            }
        }
}

// ---------------- CLS-row attention scores ----------------
__global__ __launch_bounds__(256)
void ppt_cls_scores(const unsigned short* __restrict__ Qb,
                    const unsigned short* __restrict__ Kb,
                    const unsigned short* __restrict__ Vb,
                    float* __restrict__ scores) {
    __shared__ float sig[NTOK];
    __shared__ float sarr[NTOK];
    __shared__ float q0[HD];
    __shared__ float red[256];
    const int b = blockIdx.x, tid = threadIdx.x;
    for (int j = tid; j < NTOK; j += 256) sig[j] = 0.f;
    __syncthreads();

    for (int h = 0; h < HEADS; ++h) {
        const unsigned short* Qp = Qb + (size_t)((b * HEADS + h) * NP) * HD;
        const unsigned short* Kp = Kb + (size_t)((b * HEADS + h) * NP) * HD;
        if (tid < HD) q0[tid] = bf2f(Qp[tid]);
        __syncthreads();
        float lmax = -1e30f;
        for (int j = tid; j < NTOK; j += 256) {
            float s = 0.f;
            for (int d = 0; d < HD; ++d) s += q0[d] * bf2f(Kp[(size_t)j * HD + d]);
            s *= SCALEF;
            sarr[j] = s;
            lmax = fmaxf(lmax, s);
        }
        red[tid] = lmax; __syncthreads();
        for (int s = 128; s > 0; s >>= 1) {
            if (tid < s) red[tid] = fmaxf(red[tid], red[tid + s]);
            __syncthreads();
        }
        const float M = red[0]; __syncthreads();
        float lsum = 0.f;
        for (int j = tid; j < NTOK; j += 256) {
            const float e = __expf(sarr[j] - M);
            sarr[j] = e;
            lsum += e;
        }
        red[tid] = lsum; __syncthreads();
        for (int s = 128; s > 0; s >>= 1) {
            if (tid < s) red[tid] += red[tid + s];
            __syncthreads();
        }
        const float S = red[0]; __syncthreads();
        for (int j = tid; j < NTOK; j += 256) sig[j] += sarr[j] / S;
        __syncthreads();
    }
    for (int j = tid; j < NTOK; j += 256) {
        float acc = 0.f;
        for (int h = 0; h < HEADS; ++h) {
            const unsigned short* Vp =
                Vb + (size_t)((b * HEADS + h) * NP + j) * HD;
            for (int d = 0; d < HD; ++d) {
                const float v = bf2f(Vp[d]);
                acc += v * v;
            }
        }
        sig[j] *= sqrtf(acc);
    }
    __syncthreads();
    float ls = 0.f;
    for (int j = tid; j < NTOK; j += 256)
        if (j >= 1) ls += sig[j];
    red[tid] = ls; __syncthreads();
    for (int s = 128; s > 0; s >>= 1) {
        if (tid < s) red[tid] += red[tid + s];
        __syncthreads();
    }
    const float T = red[0]; __syncthreads();
    for (int j = tid; j < NTOK; j += 256)
        scores[b * NTOK + j] = (j == 0) ? CLSCST : sig[j] / T;
}

// ---------------- projection GEMM over padded M (9728 x 768 x 768) ----------
__global__ __launch_bounds__(256)
void ppt_proj_gemm(const unsigned short* __restrict__ ab,
                   const unsigned short* __restrict__ wb,
                   const float* __restrict__ bias,
                   float* __restrict__ out) {
    __shared__ unsigned short As[2][128 * 40];
    __shared__ unsigned short Bs[2][64 * 40];
    const int t    = threadIdx.x;
    const int lane = t & 31;
    const int wave = t >> 5;
    const int half = lane >> 4;
    const int l15  = lane & 15;
    const int m0   = blockIdx.x * 128;
    const int n0   = blockIdx.y * 64;
    const int mw   = (wave & 3) * 32;
    const int nw   = (wave >> 2) * 32;

    const v8f vzero = {0.f, 0.f, 0.f, 0.f, 0.f, 0.f, 0.f, 0.f};
    v8f acc[2][2];
#pragma unroll
    for (int i = 0; i < 2; ++i)
#pragma unroll
        for (int j = 0; j < 2; ++j) acc[i][j] = vzero;

#if PPT_HAS_TDM
    const i32x8 gA = tdm_desc2d(DIMC, MPAD, 32, 128, DIMC);
    const i32x8 gB = tdm_desc2d(DIMC, DIMC, 32, 64, DIMC);
    const unsigned long long baseA =
        (unsigned long long)(uintptr_t)ab + (unsigned long long)m0 * DIMC * 2;
    const unsigned long long baseB =
        (unsigned long long)(uintptr_t)wb + (unsigned long long)n0 * DIMC * 2;
    if (wave == 0) {
        tdm_issue((unsigned)(uintptr_t)&As[0][0], baseA, gA);
        tdm_issue((unsigned)(uintptr_t)&Bs[0][0], baseB, gB);
    }
#endif

    for (int kt = 0; kt < KITERS; ++kt) {
        const int cur = kt & 1;
#if PPT_HAS_TDM
        if (wave == 0) {
            if (kt + 1 < KITERS) {
                tdm_issue((unsigned)(uintptr_t)&As[cur ^ 1][0],
                          baseA + (unsigned long long)(kt + 1) * 64, gA);
                tdm_issue((unsigned)(uintptr_t)&Bs[cur ^ 1][0],
                          baseB + (unsigned long long)(kt + 1) * 64, gB);
                __builtin_amdgcn_s_wait_tensorcnt(2);
            } else {
                __builtin_amdgcn_s_wait_tensorcnt(0);
            }
        }
        __syncthreads();
#else
        {
            const int k0 = kt * 32;
            const int r = t >> 2, c = (t & 3) * 8;
#pragma unroll
            for (int rr = 0; rr < 2; ++rr) {
                const int row = m0 + r + rr * 64;
                *(uint4*)(&As[cur][(r + rr * 64) * 40 + c]) =
                    *(const uint4*)(ab + (size_t)row * DIMC + k0 + c);
            }
            *(uint4*)(&Bs[cur][r * 40 + c]) =
                *(const uint4*)(wb + (size_t)(n0 + r) * DIMC + k0 + c);
        }
        __syncthreads();
#endif
        FragB16 a[2], b[2];
        const int kb  = half * 8;
        const int kb2 = half * 16;
#pragma unroll
        for (int i = 0; i < 2; ++i) {
            const int m = mw + i * 16 + l15;
#pragma unroll
            for (int g = 0; g < 8; ++g) {
                const int K = kb2 + 2 * g;
                a[i].w[g] = *(const unsigned int*)(&As[cur][m * 40 + K]);
            }
            const int n = nw + i * 16 + l15;
#pragma unroll
            for (int g = 0; g < 8; ++g)
                b[i].w[g] = *(const unsigned int*)(&Bs[cur][n * 40 + kb2 + 2 * g]);
        }
#pragma unroll
        for (int i = 0; i < 2; ++i)
#pragma unroll
            for (int j = 0; j < 2; ++j)
                acc[i][j] = __builtin_amdgcn_wmma_f32_16x16x32_bf16(
                    false, a[i].v, false, b[j].v, (short)0, acc[i][j],
                    false, false);
        __syncthreads();
    }

#pragma unroll
    for (int j = 0; j < 2; ++j) {
        const int col = n0 + nw + j * 16 + l15;   // 0..767
        const float bv = bias[col];
#pragma unroll
        for (int i = 0; i < 2; ++i)
#pragma unroll
            for (int r = 0; r < 8; ++r) {
                const int row = m0 + mw + i * 16 + r + half * 8;  // < 9728
                const int bb = row / NP, nn = row % NP;
                if (nn < NTOK)
                    out[(size_t)(bb * NTOK + nn) * DIMC + col] =
                        acc[i][j][r] + bv;
            }
    }
}

// ---------------- launcher ----------------
extern "C" void kernel_launch(void* const* d_in, const int* in_sizes, int n_in,
                              void* d_out, int out_size, void* d_ws,
                              size_t ws_size, hipStream_t stream) {
    (void)in_sizes; (void)n_in; (void)out_size; (void)ws_size;
    const float* x      = (const float*)d_in[0];   // (16,577,768)
    const float* qkv_w  = (const float*)d_in[1];   // (2304,768)
    const float* proj_w = (const float*)d_in[2];   // (768,768)
    const float* proj_b = (const float*)d_in[3];   // (768,)
    float* out    = (float*)d_out;                        // (16,577,768)
    float* scores = (float*)d_out + (size_t)MROWS * DIMC; // (16,577)

    const size_t nX  = (size_t)MROWS * DIMC;   // 7,090,176
    const size_t nQW = (size_t)QKVC * DIMC;    // 1,769,472
    const size_t nPW = (size_t)DIMC * DIMC;    //   589,824
    const size_t nHB = (size_t)BATCH * HEADS * NP * HD;  // per Q/K/V buffer
    const size_t nAO = (size_t)MPAD * DIMC;

    char* ws = (char*)d_ws;
    size_t off = 0;
    auto take = [&](size_t bytes) {
        char* p = ws + off;
        off += (bytes + 255) & ~(size_t)255;
        return p;
    };
    unsigned short* xb  = (unsigned short*)take(nX * 2);
    unsigned short* qwb = (unsigned short*)take(nQW * 2);
    unsigned short* pwb = (unsigned short*)take(nPW * 2);
    unsigned short* Qb  = (unsigned short*)take(nHB * 2);
    unsigned short* Kb  = (unsigned short*)take(nHB * 2);
    unsigned short* Vb  = (unsigned short*)take(nHB * 2);
    unsigned short* AOb = (unsigned short*)take(nAO * 2);

    // 1) fp32 -> bf16 conversions (element counts all divisible by 4)
    ppt_cvt_bf16<<<2048, 256, 0, stream>>>(x, xb, (int)(nX / 4));
    ppt_cvt_bf16<<<1024, 256, 0, stream>>>(qkv_w, qwb, (int)(nQW / 4));
    ppt_cvt_bf16<<<512, 256, 0, stream>>>(proj_w, pwb, (int)(nPW / 4));

    // 2) QKV projection GEMM (TDM-fed, double buffered)
    ppt_qkv_gemm<<<dim3((MROWS + 127) / 128, QKVC / 64), 256, 0, stream>>>(
        xb, qwb, Qb, Kb, Vb);

    // 3) fused flash attention (37 q-tiles x 192 head-batches)
    ppt_flash_attn<<<dim3((NTOK + 15) / 16, BATCH * HEADS), 32, 0, stream>>>(
        Qb, Kb, Vb, AOb);

    // 4) CLS-row token scores
    ppt_cls_scores<<<BATCH, 256, 0, stream>>>(Qb, Kb, Vb, scores);

    // 5) output projection GEMM (+bias) over padded M
    ppt_proj_gemm<<<dim3(MPAD / 128, DIMC / 64), 256, 0, stream>>>(
        AOb, pwb, proj_b, out);
}
